// GCN_MLP_Model_29051158790850
// MI455X (gfx1250) — compile-verified, matrix operand order checked by reading, not run on hardware
//
#include <hip/hip_runtime.h>
#include <hip/hip_bf16.h>

// Problem constants (from reference)
#define N_NODES 100000
#define N_EDGES 1600000
#define IN_CH   256
#define HID     128
#define OUT_CH  64

// ---- CDNA5 WMMA types -------------------------------------------------------
typedef __attribute__((ext_vector_type(16))) __bf16 v16bf;
typedef __attribute__((ext_vector_type(8)))  float  v8f;

union BF16Frag {
  v16bf bf;
  unsigned int u[8];
};

// fp32 -> bf16 pair, round-to-nearest-even, packed into one dword
__device__ __forceinline__ unsigned int pack_bf16(float lo, float hi) {
  unsigned int a = __float_as_uint(lo);
  unsigned int b = __float_as_uint(hi);
  a += 0x7FFFu + ((a >> 16) & 1u);
  b += 0x7FFFu + ((b >> 16) & 1u);
  return (a >> 16) | (b & 0xFFFF0000u);
}

// ---- GEMM: C[M,N] = A[M,K] @ W[K,N], fp32 in/out, bf16 WMMA inside ----------
// 256 threads = 8 wave32; each wave computes a 16-row strip x full N.
// W is converted once per workgroup into LDS, stored transposed ([N][K] bf16)
// so each WMMA B half-chunk is a single ds_load_b128.
template <int K, int N>
__global__ __launch_bounds__(256) void gemm_bf16_wmma(
    const float* __restrict__ A, const float* __restrict__ W,
    float* __restrict__ C, int M) {
  constexpr int NT = N / 16;   // 16x16 n-tiles per wave
  constexpr int KS = K / 32;   // k-steps of 32 (bf16 WMMA depth)
  constexpr int KP = K / 2;    // dwords (bf16 pairs) per transposed-W row
  __shared__ unsigned int s_w[N * KP];

  const int tid = threadIdx.x;

  // Stage W^T into LDS as bf16 pairs: s_w[n][kk] = {W[2kk][n], W[2kk+1][n]}
  for (int idx = tid; idx < N * KP; idx += 256) {
    const int n  = idx / KP;
    const int kk = idx - n * KP;
    const float w0 = W[(2 * kk)     * N + n];
    const float w1 = W[(2 * kk + 1) * N + n];
    s_w[n * KP + kk] = pack_bf16(w0, w1);
  }
  __syncthreads();

  const int lane  = tid & 31;
  const int wave  = tid >> 5;
  const int half  = lane >> 4;                 // K half-select (A) / M half (C)
  const int mbase = blockIdx.x * 128 + wave * 16;
  const int row   = mbase + (lane & 15);       // A row held by this lane
  const bool rv   = row < M;
  const float* arow = A + (size_t)row * K;

  v8f acc[NT] = {};

#pragma unroll
  for (int s = 0; s < KS; ++s) {
    const int kb = s * 32 + half * 8;
    // A fragment: 16-bit A layout = K chunks [kb,kb+8) and [kb+16,kb+24)
    BF16Frag af;
    if (rv) {
      const float4 f0 = *(const float4*)(arow + kb);
      const float4 f1 = *(const float4*)(arow + kb + 4);
      const float4 f2 = *(const float4*)(arow + kb + 16);
      const float4 f3 = *(const float4*)(arow + kb + 20);
      af.u[0] = pack_bf16(f0.x, f0.y);
      af.u[1] = pack_bf16(f0.z, f0.w);
      af.u[2] = pack_bf16(f1.x, f1.y);
      af.u[3] = pack_bf16(f1.z, f1.w);
      af.u[4] = pack_bf16(f2.x, f2.y);
      af.u[5] = pack_bf16(f2.z, f2.w);
      af.u[6] = pack_bf16(f3.x, f3.y);
      af.u[7] = pack_bf16(f3.z, f3.w);
    } else {
#pragma unroll
      for (int i = 0; i < 8; ++i) af.u[i] = 0u;
    }
#pragma unroll
    for (int j = 0; j < NT; ++j) {
      const int n = j * 16 + (lane & 15);
      const unsigned int* bp = &s_w[n * KP + s * 16 + half * 4];
      const uint4 b0 = *(const uint4*)(bp);       // K chunk [kb, kb+8)
      const uint4 b1 = *(const uint4*)(bp + 8);   // K chunk [kb+16, kb+24)
      BF16Frag bfr;
      bfr.u[0] = b0.x; bfr.u[1] = b0.y; bfr.u[2] = b0.z; bfr.u[3] = b0.w;
      bfr.u[4] = b1.x; bfr.u[5] = b1.y; bfr.u[6] = b1.z; bfr.u[7] = b1.w;
      acc[j] = __builtin_amdgcn_wmma_f32_16x16x32_bf16(
          false, af.bf, false, bfr.bf, (short)0, acc[j], false, false);
    }
  }

  // Store D: VGPR r holds M = r + 8*(lane/16), N = lane%16 within the tile.
#pragma unroll
  for (int j = 0; j < NT; ++j) {
    const int col = j * 16 + (lane & 15);
#pragma unroll
    for (int r = 0; r < 8; ++r) {
      const int m = mbase + half * 8 + r;
      if (m < M) C[(size_t)m * N + col] = acc[j][r];
    }
  }
}

// ---- out1 := broadcast(bias) (scatter-add then accumulates on top) ----------
__global__ __launch_bounds__(256) void init_bias_kernel(
    float* __restrict__ out1, const float* __restrict__ b) {
  const size_t idx = (size_t)blockIdx.x * 256 + threadIdx.x;
  if (idx < (size_t)N_NODES * HID) out1[idx] = b[idx & (HID - 1)];
}

// ---- edge scatter-add: one wave32 per edge, float4 gather + atomic adds -----
__global__ __launch_bounds__(256) void scatter_add_kernel(
    const float* __restrict__ h, const int* __restrict__ ei,
    float* __restrict__ out1) {
  const int e = (blockIdx.x * 256 + (int)threadIdx.x) >> 5;
  if (e >= N_EDGES) return;
  const int lane = threadIdx.x & 31;
  const int src = ei[e];
  const int dst = ei[N_EDGES + e];
  const float4 v = *(const float4*)(h + (size_t)src * HID + lane * 4);
  float* o = out1 + (size_t)dst * HID + lane * 4;
  atomicAdd(o + 0, v.x);
  atomicAdd(o + 1, v.y);
  atomicAdd(o + 2, v.z);
  atomicAdd(o + 3, v.w);
}

extern "C" void kernel_launch(void* const* d_in, const int* in_sizes, int n_in,
                              void* d_out, int out_size, void* d_ws, size_t ws_size,
                              hipStream_t stream) {
  const float* x     = (const float*)d_in[0];   // [N_NODES, IN_CH]
  const int*   ei    = (const int*)  d_in[1];   // [2, N_EDGES]
  const float* W_gcn = (const float*)d_in[2];   // [IN_CH, HID]
  const float* b_gcn = (const float*)d_in[3];   // [HID]
  const float* W1    = (const float*)d_in[4];   // [HID, HID]
  const float* W2    = (const float*)d_in[5];   // [HID, OUT_CH]

  float* out  = (float*)d_out;
  float* out1 = out;                                   // [N_NODES, HID]
  float* out2 = out + (size_t)N_NODES * HID;           // [N_NODES, HID]
  float* out3 = out + (size_t)N_NODES * HID * 2;       // [N_NODES, OUT_CH]

  // Scratch for pre-aggregation features h = x @ W_gcn.
  // Fall back to the out2 slab (it's overwritten by the real out2 later,
  // and h is dead by then) if the workspace is too small.
  const size_t h_bytes = (size_t)N_NODES * HID * sizeof(float);
  float* h = (ws_size >= h_bytes) ? (float*)d_ws : out2;

  const int mblocks = (N_NODES + 127) / 128;

  // 1) out1 = bias (scatter-add target)
  {
    const size_t tot = (size_t)N_NODES * HID;
    init_bias_kernel<<<(unsigned)((tot + 255) / 256), 256, 0, stream>>>(out1, b_gcn);
  }
  // 2) h = x @ W_gcn   (bf16 WMMA, fp32 accumulate)
  gemm_bf16_wmma<IN_CH, HID><<<mblocks, 256, 0, stream>>>(x, W_gcn, h, N_NODES);
  // 3) out1 += segment_sum(h[src] -> dst)
  scatter_add_kernel<<<(N_EDGES * 32 + 255) / 256, 256, 0, stream>>>(h, ei, out1);
  // 4) out2 = out1 @ W1
  gemm_bf16_wmma<HID, HID><<<mblocks, 256, 0, stream>>>(out1, W1, out2, N_NODES);
  // 5) out3 = out2 @ W2
  gemm_bf16_wmma<HID, OUT_CH><<<mblocks, 256, 0, stream>>>(out2, W2, out3, N_NODES);
}
